// CWN_34471407517841
// MI455X (gfx1250) — compile-verified, matrix operand order checked by reading, not run on hardware
//
#include <hip/hip_runtime.h>
#include <hip/hip_bf16.h>
#include <cstdint>
#include <cstddef>

#define H 128

typedef __bf16 bf16_t;
typedef __attribute__((ext_vector_type(16))) __bf16 v16bf;
typedef __attribute__((ext_vector_type(8)))  float  v8f;

union ABFrag { v16bf v; uint4 u[2]; };
union BfQuad { bf16_t h[4]; uint2 u; };   // 4 packed bf16 = 8 bytes

static inline int cdiv_h(int a, int b) { return (a + b - 1) / b; }

__device__ __forceinline__ float elu_f(float x) {
  return (x > 0.0f) ? x : (__expf(x) - 1.0f);
}

// ---------------------------------------------------------------------------
// Fused GEMM:  Y[N][LDY(=128|256 slice)] = act(X[N][K] @ W[K][128] + bias)
// X bf16 row-major, Wt bf16 = W transposed ([128][K]).
// Block = 256 threads (8 waves) -> 64 rows x 128 cols. Wave owns a 16-col
// tile; B fragments in registers reused over 4 row sub-tiles; A in LDS.
// LDY is a compile-time constant so row stores fold into immediate offsets.
// ---------------------------------------------------------------------------
template<int K, int ACT_ELU, int OUT_BF16, int LDY>
__global__ __launch_bounds__(256) void k_gemm(
    const bf16_t* __restrict__ X,
    const bf16_t* __restrict__ Wt,
    const float*  __restrict__ bias,
    void* __restrict__ Yv, int N)
{
  constexpr int KT  = K / 32;     // wmma k-steps
  constexpr int LDA = K + 8;      // pad 16B to kill LDS bank conflicts
  __shared__ __align__(16) bf16_t As[64 * LDA];

  const int row0 = blockIdx.x * 64;
  const int tid  = threadIdx.x;

  __builtin_prefetch(Wt, 0, 3);   // global_prefetch_b8 (weights)

  // ---- stage A tile (64 x K bf16): b128 global loads -> b128 LDS stores ----
  constexpr int VPR = K / 8;      // uint4 vectors per row
  for (int i = tid; i < 64 * VPR; i += 256) {
    int r = i >> (K == 128 ? 4 : 5);
    int c = (i - r * VPR) * 8;
    uint4 val = make_uint4(0u, 0u, 0u, 0u);
    int gr = row0 + r;
    if (gr < N) val = *(const uint4*)(X + (size_t)gr * K + c);
    *(uint4*)(&As[r * LDA + c]) = val;
  }
  __syncthreads();

  const int wave = tid >> 5;
  const int lane = tid & 31;
  const int lm   = lane & 15;     // M (A) / N (B,C,D) within tile
  const int lh   = lane >> 4;     // half select (K range)
  const int n0   = wave * 16;     // this wave's column tile

  // ---- B fragments: Wt[n][k] contiguous in k -> two b128 loads each ----
  ABFrag bfr[KT];
#pragma unroll
  for (int kt = 0; kt < KT; ++kt) {
    const bf16_t* wp = Wt + (size_t)(n0 + lm) * K + kt * 32 + lh * 8;
    bfr[kt].u[0] = *(const uint4*)(wp);
    bfr[kt].u[1] = *(const uint4*)(wp + 16);
  }
  const float bval = bias ? bias[n0 + lm] : 0.0f;
  const bool  full = (row0 + 64 <= N);   // wave-uniform fast path

#pragma unroll
  for (int rt = 0; rt < 4; ++rt) {
    v8f acc = {};
#pragma unroll
    for (int kt = 0; kt < KT; ++kt) {
      ABFrag a;
      const bf16_t* ap = &As[(rt * 16 + lm) * LDA + kt * 32 + lh * 8];
      a.u[0] = *(const uint4*)(ap);
      a.u[1] = *(const uint4*)(ap + 16);
      acc = __builtin_amdgcn_wmma_f32_16x16x32_bf16(
          /*neg_a=*/false, a.v, /*neg_b=*/false, bfr[kt].v,
          /*c_mod=*/(short)0, acc, /*reuse_a=*/false, /*reuse_b=*/false);
    }
    // ---- epilogue: bias + optional ELU computed first ----
    float ys[8];
#pragma unroll
    for (int e = 0; e < 8; ++e) {
      float y = acc[e] + bval;
      if (ACT_ELU) y = elu_f(y);
      ys[e] = y;
    }
    // rows rbase..rbase+7, one column per lane; LDY compile-time so the
    // 8 stores use immediate offsets off a single base address.
    const int rbase = row0 + rt * 16 + 8 * lh;
    if (OUT_BF16) {
      bf16_t* p = (bf16_t*)Yv + (size_t)rbase * LDY + n0 + lm;
      if (full) {
#pragma unroll
        for (int e = 0; e < 8; ++e) p[e * LDY] = (bf16_t)ys[e];
      } else {
#pragma unroll
        for (int e = 0; e < 8; ++e) if (rbase + e < N) p[e * LDY] = (bf16_t)ys[e];
      }
    } else {
      float* p = (float*)Yv + (size_t)rbase * LDY + n0 + lm;
      if (full) {
#pragma unroll
        for (int e = 0; e < 8; ++e) p[e * LDY] = ys[e];
      } else {
#pragma unroll
        for (int e = 0; e < 8; ++e) if (rbase + e < N) p[e * LDY] = ys[e];
      }
    }
  }
}

// ---------------------------------------------------------------------------
// COO SpMM scatter: Y[rows[e]] += vals[e] * X[cols[e]]  (H=128, f32)
// One wave per edge: coalesced 512B row gather, float fadd atomics scatter.
// ---------------------------------------------------------------------------
__global__ __launch_bounds__(256) void k_spmm(
    const int* __restrict__ rows, const int* __restrict__ cols,
    const float* __restrict__ vals, const float* __restrict__ X,
    float* __restrict__ Y, int nnz)
{
  int e = blockIdx.x * 8 + (threadIdx.x >> 5);
  if (e >= nnz) return;
  int lane = threadIdx.x & 31;
  int rs = cols[e];
  int rd = rows[e];
  float v = vals[e];
  float4 xv = ((const float4*)(X + (size_t)rs * H))[lane];
  float* yp = Y + (size_t)rd * H + lane * 4;
  __hip_atomic_fetch_add(yp + 0, v * xv.x, __ATOMIC_RELAXED, __HIP_MEMORY_SCOPE_AGENT);
  __hip_atomic_fetch_add(yp + 1, v * xv.y, __ATOMIC_RELAXED, __HIP_MEMORY_SCOPE_AGENT);
  __hip_atomic_fetch_add(yp + 2, v * xv.z, __ATOMIC_RELAXED, __HIP_MEMORY_SCOPE_AGENT);
  __hip_atomic_fetch_add(yp + 3, v * xv.w, __ATOMIC_RELAXED, __HIP_MEMORY_SCOPE_AGENT);
}

// ---------------------------------------------------------------------------
// BatchNorm column statistics: stats[0..127]=sum, stats[128..255]=sumsq
// Shared-memory (ds_add_f32) partial reduction + global fadd atomics.
// ---------------------------------------------------------------------------
__global__ __launch_bounds__(256) void k_bn_stats(
    const float* __restrict__ X, float* __restrict__ stats, int Nrows)
{
  __shared__ float ssum[H];
  __shared__ float ssq[H];
  int tid = threadIdx.x;
  if (tid < H) { ssum[tid] = 0.0f; ssq[tid] = 0.0f; }
  __syncthreads();
  int c = tid & (H - 1);
  float ls = 0.0f, lq = 0.0f;
  for (int r = blockIdx.x * 2 + (tid >> 7); r < Nrows; r += gridDim.x * 2) {
    float v = X[(size_t)r * H + c];
    ls += v;
    lq += v * v;
  }
  atomicAdd(&ssum[c], ls);
  atomicAdd(&ssq[c], lq);
  __syncthreads();
  if (tid < H) {
    __hip_atomic_fetch_add(&stats[tid],     ssum[tid], __ATOMIC_RELAXED, __HIP_MEMORY_SCOPE_AGENT);
    __hip_atomic_fetch_add(&stats[H + tid], ssq[tid],  __ATOMIC_RELAXED, __HIP_MEMORY_SCOPE_AGENT);
  }
}

// BN normalize + ReLU, emit bf16 (vec4: float4 in, packed 4xbf16 out).
__global__ __launch_bounds__(256) void k_bn_relu_bf16(
    const float* __restrict__ X, const float* __restrict__ stats,
    const float* __restrict__ g, const float* __restrict__ be,
    bf16_t* __restrict__ Y, int nvec, float invN)   // nvec = total/4
{
  for (int i = blockIdx.x * 256 + threadIdx.x; i < nvec; i += gridDim.x * 256) {
    int c4 = (i & ((H / 4) - 1)) * 4;
    float4 xv = ((const float4*)X)[i];
    BfQuad q;
#pragma unroll
    for (int j = 0; j < 4; ++j) {
      int c = c4 + j;
      float m  = stats[c] * invN;
      float v  = stats[H + c] * invN - m * m;
      float sc = g[c] * rsqrtf(v + 1e-5f);
      float x  = (&xv.x)[j];
      (void)0;
      float y  = sc * (x - m) + be[c];
      q.h[j] = (bf16_t)fmaxf(y, 0.0f);
    }
    ((uint2*)Y)[i] = q.u;
  }
}

// out[r*LDO + coff + c] = bf16((1+eps)*x1 + (elu?)(s)) — fused residual,
// vec4 loads, packed 4xbf16 strided store (builds concat halves directly).
template<int DO_ELU, int LDO>
__global__ __launch_bounds__(256) void k_resid_bf16(
    const float* __restrict__ x1, const float* __restrict__ s,
    const float* __restrict__ epsp, bf16_t* __restrict__ out,
    int nvec, int coff)   // nvec = N*H/4
{
  const float c1 = 1.0f + epsp[0];
  for (int i = blockIdx.x * 256 + threadIdx.x; i < nvec; i += gridDim.x * 256) {
    int r  = i >> 5;                    // H/4 = 32 vec4 per row
    int c4 = (i & 31) * 4;
    float4 xv = ((const float4*)x1)[i];
    float4 sv = ((const float4*)s)[i];
    BfQuad q;
    float svs[4] = {sv.x, sv.y, sv.z, sv.w};
    float xvs[4] = {xv.x, xv.y, xv.z, xv.w};
#pragma unroll
    for (int j = 0; j < 4; ++j) {
      float t = svs[j];
      if (DO_ELU) t = elu_f(t);
      q.h[j] = (bf16_t)(c1 * xvs[j] + t);
    }
    *(uint2*)(&out[(size_t)r * LDO + coff + c4]) = q.u;
  }
}

// f32 -> bf16 elementwise (vec4)
__global__ __launch_bounds__(256) void k_cvt_bf16(
    const float* __restrict__ X, bf16_t* __restrict__ Y, size_t nvec)
{
  for (size_t i = blockIdx.x * 256ull + threadIdx.x; i < nvec;
       i += (size_t)gridDim.x * 256ull) {
    float4 xv = ((const float4*)X)[i];
    BfQuad q;
    q.h[0] = (bf16_t)xv.x; q.h[1] = (bf16_t)xv.y;
    q.h[2] = (bf16_t)xv.z; q.h[3] = (bf16_t)xv.w;
    ((uint2*)Y)[i] = q.u;
  }
}

// W[K][128] f32 -> Wt[128][K] bf16 (transpose + convert, once per weight)
__global__ __launch_bounds__(256) void k_wt_cvt(
    const float* __restrict__ W, bf16_t* __restrict__ Wt, int K)
{
  int total = K * H;
  for (int i = blockIdx.x * 256 + threadIdx.x; i < total; i += gridDim.x * 256) {
    int k = i >> 7;
    int n = i & (H - 1);
    Wt[(size_t)n * K + k] = (bf16_t)W[i];
  }
}

// ---------------------------------------------------------------------------
extern "C" void kernel_launch(void* const* d_in, const int* in_sizes, int n_in,
                              void* d_out, int out_size, void* d_ws, size_t ws_size,
                              hipStream_t stream)
{
  (void)n_in; (void)out_size; (void)ws_size;

  const int N0 = in_sizes[0] / H;
  const int N1 = in_sizes[1] / H;
  const int N2 = in_sizes[2] / H;
  const int NNZ11 = in_sizes[3];
  const int NNZ21 = in_sizes[6];
  const int NNZ01 = in_sizes[9];
  const size_t N1H = (size_t)N1 * H;
  const int N1V = (int)(N1H / 4);       // vec4 count for N1 x H tensors

  const float* x0_in = (const float*)d_in[0];
  const float* x1_in = (const float*)d_in[1];
  const float* x2_in = (const float*)d_in[2];
  const int*   r11 = (const int*)d_in[3];
  const int*   c11 = (const int*)d_in[4];
  const float* v11 = (const float*)d_in[5];
  const int*   r21 = (const int*)d_in[6];
  const int*   c21 = (const int*)d_in[7];
  const float* v21 = (const float*)d_in[8];
  const int*   r01 = (const int*)d_in[9];
  const int*   c01 = (const int*)d_in[10];
  const float* v01 = (const float*)d_in[11];
  const float* p0w = (const float*)d_in[12];
  const float* p0b = (const float*)d_in[13];
  const float* p1w = (const float*)d_in[14];
  const float* p1b = (const float*)d_in[15];
  const float* p2w = (const float*)d_in[16];
  const float* p2b = (const float*)d_in[17];

  // ---- workspace bump allocator ----
  uint8_t* wsp = (uint8_t*)d_ws;
  size_t off = 0;
  auto alloc = [&](size_t bytes) -> void* {
    void* p = wsp + off;
    off += (bytes + 255) & ~(size_t)255;
    return p;
  };

  int NMAX = N1 > N0 ? N1 : N0; if (N2 > NMAX) NMAX = N2;

  float* f_x1a = (float*)alloc(N1H * 4);              // layer state A
  float* f_x1b = (float*)alloc(N1H * 4);              // layer state B
  float* f_s   = (float*)alloc(N1H * 4);              // spmm accumulator
  float* f_tA  = (float*)alloc((size_t)NMAX * H * 4); // pre-spmm / fc_out
  float* f_tB  = (float*)alloc(N1H * 4);              // MLP hidden (pre-BN)
  float* stats = (float*)alloc(2 * H * 4);            // BN sum/sumsq
  bf16_t* b_x0  = (bf16_t*)alloc((size_t)N0 * H * 2); // x0 proj (bf16, persistent)
  bf16_t* b_x2  = (bf16_t*)alloc((size_t)N2 * H * 2); // x2 proj (bf16, persistent)
  bf16_t* b_x1  = (bf16_t*)alloc(N1H * 2);            // bf16 of current x1
  bf16_t* b_cat = (bf16_t*)alloc(N1H * 2 * 2);        // [N1][256] concat
  bf16_t* b_h   = (bf16_t*)alloc(N1H * 2);            // post-BN hidden
  bf16_t* b_t   = (bf16_t*)alloc(N1H * 2);            // arrow-in / agg (bf16)

  bf16_t* wt_p0 = (bf16_t*)alloc(H * H * 2);
  bf16_t* wt_p1 = (bf16_t*)alloc(H * H * 2);
  bf16_t* wt_p2 = (bf16_t*)alloc(H * H * 2);
  bf16_t *wt_fc11[2], *wt_fc21[2], *wt_fcm1[2], *wt_fcm2[2], *wt_sc[2];
  bf16_t *wt_ar1[2], *wt_ar2[2], *wt_m1[2], *wt_m2[2], *wt_upd[2];
  for (int L = 0; L < 2; ++L) {
    wt_fc11[L] = (bf16_t*)alloc(H * H * 2);
    wt_fc21[L] = (bf16_t*)alloc(H * H * 2);
    wt_fcm1[L] = (bf16_t*)alloc(2 * H * H * 2);
    wt_fcm2[L] = (bf16_t*)alloc(H * H * 2);
    wt_sc[L]   = (bf16_t*)alloc(H * H * 2);
    wt_ar1[L]  = (bf16_t*)alloc(H * H * 2);
    wt_ar2[L]  = (bf16_t*)alloc(H * H * 2);
    wt_m1[L]   = (bf16_t*)alloc(2 * H * H * 2);
    wt_m2[L]   = (bf16_t*)alloc(H * H * 2);
    wt_upd[L]  = (bf16_t*)alloc(H * H * 2);
  }

  auto wtc = [&](const float* W, bf16_t* Wt, int K) {
    k_wt_cvt<<<cdiv_h(K * H, 256), 256, 0, stream>>>(W, Wt, K);
  };
  auto cvt = [&](const float* X, bf16_t* Y, size_t n) {
    k_cvt_bf16<<<2048, 256, 0, stream>>>(X, Y, n / 4);
  };

  // ---- one-time weight transpose/convert ----
  wtc(p0w, wt_p0, H); wtc(p1w, wt_p1, H); wtc(p2w, wt_p2, H);
  for (int L = 0; L < 2; ++L) {
    const int pb = 18 + 25 * L;
    wtc((const float*)d_in[pb + 2],  wt_fc11[L], H);
    wtc((const float*)d_in[pb + 3],  wt_fc21[L], H);
    wtc((const float*)d_in[pb + 4],  wt_fcm1[L], 2 * H);
    wtc((const float*)d_in[pb + 8],  wt_fcm2[L], H);
    wtc((const float*)d_in[pb + 10], wt_sc[L],   H);
    wtc((const float*)d_in[pb + 11], wt_ar1[L],  H);
    wtc((const float*)d_in[pb + 15], wt_ar2[L],  H);
    wtc((const float*)d_in[pb + 17], wt_m1[L],   2 * H);
    wtc((const float*)d_in[pb + 21], wt_m2[L],   H);
    wtc((const float*)d_in[pb + 23], wt_upd[L],  H);
  }

  const int g0 = cdiv_h(N0, 64), g1 = cdiv_h(N1, 64), g2 = cdiv_h(N2, 64);
  const float invN1 = 1.0f / (float)N1;

  // ---- initial projections ----
  cvt(x0_in, b_x0, (size_t)N0 * H);
  k_gemm<128,0,1,128><<<g0, 256, 0, stream>>>(b_x0, wt_p0, p0b, (void*)b_x0, N0);
  cvt(x2_in, b_x2, (size_t)N2 * H);
  k_gemm<128,0,1,128><<<g2, 256, 0, stream>>>(b_x2, wt_p2, p2b, (void*)b_x2, N2);
  cvt(x1_in, b_x1, N1H);
  k_gemm<128,0,0,128><<<g1, 256, 0, stream>>>(b_x1, wt_p1, p1b, (void*)f_x1a, N1);

  // ---- CWN layers ----
  float* curf = f_x1a;
  for (int L = 0; L < 2; ++L) {
    const int pb = 18 + 25 * L;
    const float* eps_l  = (const float*)d_in[pb + 0];
    const float* eps_fc = (const float*)d_in[pb + 1];
    const float* fcm_b1 = (const float*)d_in[pb + 5];
    const float* fcm_g  = (const float*)d_in[pb + 6];
    const float* fcm_be = (const float*)d_in[pb + 7];
    const float* fcm_b2 = (const float*)d_in[pb + 9];
    const float* ar_b1  = (const float*)d_in[pb + 12];
    const float* ar_g   = (const float*)d_in[pb + 13];
    const float* ar_be  = (const float*)d_in[pb + 14];
    const float* ar_b2  = (const float*)d_in[pb + 16];
    const float* m_b1   = (const float*)d_in[pb + 18];
    const float* m_g    = (const float*)d_in[pb + 19];
    const float* m_be   = (const float*)d_in[pb + 20];
    const float* m_b2   = (const float*)d_in[pb + 22];
    const float* upd_b  = (const float*)d_in[pb + 24];
    float* nxt = (L == 1) ? (float*)d_out : f_x1b;

    cvt(curf, b_x1, N1H);

    // x_up = (1+eps_fc)*x1 + elu(spmm11(x1 @ fc_w11))  -> cat[:, :128]
    k_gemm<128,0,0,128><<<g1, 256, 0, stream>>>(b_x1, wt_fc11[L], nullptr, (void*)f_tA, N1);
    hipMemsetAsync(f_s, 0, N1H * 4, stream);
    k_spmm<<<cdiv_h(NNZ11, 8), 256, 0, stream>>>(r11, c11, v11, f_tA, f_s, NNZ11);
    k_resid_bf16<1, 2 * H><<<2048, 256, 0, stream>>>(curf, f_s, eps_fc, b_cat, N1V, 0);

    // x_cob = (1+eps_fc)*x1 + elu(spmm21(x2 @ fc_w21)) -> cat[:, 128:]
    k_gemm<128,0,0,128><<<g2, 256, 0, stream>>>(b_x2, wt_fc21[L], nullptr, (void*)f_tA, N2);
    hipMemsetAsync(f_s, 0, N1H * 4, stream);
    k_spmm<<<cdiv_h(NNZ21, 8), 256, 0, stream>>>(r21, c21, v21, f_tA, f_s, NNZ21);
    k_resid_bf16<1, 2 * H><<<2048, 256, 0, stream>>>(curf, f_s, eps_fc, b_cat, N1V, H);

    // fc_out = MLP_256(cat)
    k_gemm<256,0,0,128><<<g1, 256, 0, stream>>>(b_cat, wt_fcm1[L], fcm_b1, (void*)f_tB, N1);
    hipMemsetAsync(stats, 0, 2 * H * 4, stream);
    k_bn_stats<<<1024, 256, 0, stream>>>(f_tB, stats, N1);
    k_bn_relu_bf16<<<2048, 256, 0, stream>>>(f_tB, stats, fcm_g, fcm_be, b_h, N1V, invN1);
    k_gemm<128,0,0,128><<<g1, 256, 0, stream>>>(b_h, wt_fcm2[L], fcm_b2, (void*)f_tA, N1);

    // xc11 = arrowMLP((1+eps_l)*x1 + fc_out)  -> cat[:, 128:] (bf16)
    k_resid_bf16<0, H><<<2048, 256, 0, stream>>>(curf, f_tA, eps_l, b_t, N1V, 0);
    k_gemm<128,0,0,128><<<g1, 256, 0, stream>>>(b_t, wt_ar1[L], ar_b1, (void*)f_tB, N1);
    hipMemsetAsync(stats, 0, 2 * H * 4, stream);
    k_bn_stats<<<1024, 256, 0, stream>>>(f_tB, stats, N1);
    k_bn_relu_bf16<<<2048, 256, 0, stream>>>(f_tB, stats, ar_g, ar_be, b_h, N1V, invN1);
    k_gemm<128,0,1,256><<<g1, 256, 0, stream>>>(b_h, wt_ar2[L], ar_b2, (void*)(b_cat + H), N1);

    // xc01 = (1+eps_l)*x1 + elu(spmm01(x0 @ sc_w))  -> cat[:, :128]
    k_gemm<128,0,0,128><<<g0, 256, 0, stream>>>(b_x0, wt_sc[L], nullptr, (void*)f_tA, N0);
    hipMemsetAsync(f_s, 0, N1H * 4, stream);
    k_spmm<<<cdiv_h(NNZ01, 8), 256, 0, stream>>>(r01, c01, v01, f_tA, f_s, NNZ01);
    k_resid_bf16<1, 2 * H><<<2048, 256, 0, stream>>>(curf, f_s, eps_l, b_cat, N1V, 0);

    // x_agg = MLP_256(cat);  x1_next = elu(x_agg @ upd_w + upd_b)
    k_gemm<256,0,0,128><<<g1, 256, 0, stream>>>(b_cat, wt_m1[L], m_b1, (void*)f_tB, N1);
    hipMemsetAsync(stats, 0, 2 * H * 4, stream);
    k_bn_stats<<<1024, 256, 0, stream>>>(f_tB, stats, N1);
    k_bn_relu_bf16<<<2048, 256, 0, stream>>>(f_tB, stats, m_g, m_be, b_h, N1V, invN1);
    k_gemm<128,0,1,128><<<g1, 256, 0, stream>>>(b_h, wt_m2[L], m_b2, (void*)b_t, N1);
    k_gemm<128,1,0,128><<<g1, 256, 0, stream>>>(b_t, wt_upd[L], upd_b, (void*)nxt, N1);

    curf = nxt;
  }
}
